// KiviAttentionWrapper_18614388261461
// MI455X (gfx1250) — matrix operand order repaired and checked by reading, not compiled, not run on hardware
//
#include <hip/hip_runtime.h>
#include <hip/hip_bf16.h>

// ---------------------------------------------------------------------------
// KIVI attention block for MI455X (gfx1250, wave32, WMMA + async-to-LDS).
//   hidden @ w_attn + b  -> QKV      (v_wmma_f32_16x16x32_f16, async A staging)
//   fake-quant(K), flash softmax(Q K^T + mask)/8, @V  (v_wmma, fused in LDS)
//   ctx @ w_proj + b     -> out      (v_wmma_f32_16x16x32_f16)
// ---------------------------------------------------------------------------

typedef __attribute__((ext_vector_type(16))) _Float16 v16h;
typedef __attribute__((ext_vector_type(8)))  _Float16 v8h;
typedef __attribute__((ext_vector_type(4)))  _Float16 v4h;
typedef __attribute__((ext_vector_type(8)))  float    v8f;

#define BATCH 4
#define SEQ   1024
#define EMB   1024
#define HEADS 16
#define HD    64
#define MTOT  (BATCH*SEQ)        // 4096 rows

__device__ __forceinline__ v16h wmma_pack(const v8h lo, const v8h hi) {
  v16h r;
#pragma unroll
  for (int i = 0; i < 8; ++i) { r[i] = lo[i]; r[i + 8] = hi[i]; }
  return r;
}

// Async 16B global -> LDS copy (GLOBAL_LOAD_ASYNC_TO_LDS_B128, ASYNCcnt).
// lds_off: byte offset into the group segment (low 32 bits of a generic
// pointer to a __shared__ object). gptr: per-lane 64-bit global address.
__device__ __forceinline__ void async_copy_b128(unsigned lds_off, const void* gptr) {
  asm volatile("global_load_async_to_lds_b128 %0, %1, off"
               :: "v"(lds_off), "v"(gptr)
               : "memory");
}

__device__ __forceinline__ void wait_asynccnt0() {
#if __has_builtin(__builtin_amdgcn_s_wait_asynccnt)
  __builtin_amdgcn_s_wait_asynccnt(0);
#else
  asm volatile("s_wait_asynccnt 0x0" ::: "memory");
#endif
}

// --------------------------- fp32 -> f16 convert ---------------------------
__global__ void cvt4(const float* __restrict__ src, _Float16* __restrict__ dst, int n4) {
  int i = blockIdx.x * blockDim.x + threadIdx.x;
  if (i < n4) {
    const float4 v = reinterpret_cast<const float4*>(src)[i];
    v4h h;
    h[0] = (_Float16)v.x; h[1] = (_Float16)v.y;
    h[2] = (_Float16)v.z; h[3] = (_Float16)v.w;
    reinterpret_cast<v4h*>(dst)[i] = h;
  }
}

// ------------------------- f16 GEMM, f32 accumulate ------------------------
// C[M][N] = A[M][K] @ B[K][N] + bias.
// Block 256 threads = 8 waves, tile 256(M) x 64(N); each wave owns a 32x64
// strip = 8 WMMA tiles (2 A-frags x 4 B-frags, B shared by all waves).
// Double-buffered LDS, one barrier per 32-deep K-step, A staged with
// global_load_async_to_lds_b128, next B tile prefetched into registers.
__global__ __launch_bounds__(256) void gemm_f16_wmma(
    const _Float16* __restrict__ A,
    const _Float16* __restrict__ Bw,
    const float* __restrict__ bias,
    float* __restrict__ C,
    int M, int N, int K)
{
  __shared__ __align__(16) _Float16 lA[2][256][40];   // [m][k], +8 half pad
  __shared__ __align__(16) _Float16 lBt[2][64][40];   // [n][k] transposed, +8 pad

  const int tid  = threadIdx.x;
  const int wave = tid >> 5;
  const int lane = tid & 31;
  const int ln   = lane & 15;
  const int hf   = (lane < 16) ? 0 : 1;
  const int m0   = blockIdx.y * 256;
  const int n0   = blockIdx.x * 64;
  const int wm   = wave * 32;

  const int bk   = tid >> 3;     // 0..31  (K row within tile)
  const int bnseg = tid & 7;     // 0..7   (8-half segment along N)

  v8f acc[2][4];
#pragma unroll
  for (int r2 = 0; r2 < 2; ++r2)
#pragma unroll
    for (int t = 0; t < 4; ++t)
#pragma unroll
      for (int j = 0; j < 8; ++j) acc[r2][t][j] = 0.0f;

  // ---- prologue: stage K-step 0 into buffer 0 ----
  {
    const v8h breg = *reinterpret_cast<const v8h*>(
        &Bw[(size_t)bk * N + n0 + bnseg * 8]);
#pragma unroll
    for (int i = 0; i < 4; ++i) {
      const int c = tid + i * 256;            // 0..1023
      const int row = c >> 2, seg = c & 3;
      async_copy_b128((unsigned)(size_t)&lA[0][row][seg * 8],
                      &A[(size_t)(m0 + row) * K + seg * 8]);
    }
#pragma unroll
    for (int r = 0; r < 8; ++r) lBt[0][bnseg * 8 + r][bk] = breg[r];
  }
  wait_asynccnt0();
  __syncthreads();

  int cur = 0;
  for (int k0 = 32; k0 <= K; k0 += 32) {      // K/32 iterations
    const bool has_next = (k0 < K);

    // Prefetch next tile: B into registers, A via async-to-LDS (other buffer).
    v8h breg;
    if (has_next) {
      breg = *reinterpret_cast<const v8h*>(
          &Bw[(size_t)(k0 + bk) * N + n0 + bnseg * 8]);
#pragma unroll
      for (int i = 0; i < 4; ++i) {
        const int c = tid + i * 256;
        const int row = c >> 2, seg = c & 3;
        async_copy_b128((unsigned)(size_t)&lA[cur ^ 1][row][seg * 8],
                        &A[(size_t)(m0 + row) * K + k0 + seg * 8]);
      }
    }

    // Compute on current buffer: 2 A-frags x 4 B-frags -> 8 WMMAs.
    v16h af[2];
#pragma unroll
    for (int r2 = 0; r2 < 2; ++r2) {
      const int arow = wm + r2 * 16 + ln;
      const v8h a_lo = *reinterpret_cast<const v8h*>(&lA[cur][arow][hf * 8]);
      const v8h a_hi = *reinterpret_cast<const v8h*>(&lA[cur][arow][hf * 8 + 16]);
      af[r2] = wmma_pack(a_lo, a_hi);
    }
    v16h bf[4];
#pragma unroll
    for (int t = 0; t < 4; ++t) {
      const int bn = t * 16 + ln;
      const v8h b_lo = *reinterpret_cast<const v8h*>(&lBt[cur][bn][hf * 8]);
      const v8h b_hi = *reinterpret_cast<const v8h*>(&lBt[cur][bn][hf * 8 + 16]);
      bf[t] = wmma_pack(b_lo, b_hi);
    }
#pragma unroll
    for (int r2 = 0; r2 < 2; ++r2)
#pragma unroll
      for (int t = 0; t < 4; ++t)
        acc[r2][t] = __builtin_amdgcn_wmma_f32_16x16x32_f16(
            false, af[r2], false, bf[t], (short)0, acc[r2][t], false, false);

    if (has_next) {
#pragma unroll
      for (int r = 0; r < 8; ++r) lBt[cur ^ 1][bnseg * 8 + r][bk] = breg[r];
      wait_asynccnt0();
      __syncthreads();            // single barrier per K-step
      cur ^= 1;
    }
  }

  // Epilogue: C layout row = j + 8*half, col = t*16 + ln.
#pragma unroll
  for (int r2 = 0; r2 < 2; ++r2)
#pragma unroll
    for (int t = 0; t < 4; ++t) {
      const int col = n0 + t * 16 + ln;
      const float bv = bias[col];
#pragma unroll
      for (int j = 0; j < 8; ++j) {
        const int row = m0 + wm + r2 * 16 + j + hf * 8;
        C[(size_t)row * N + col] = acc[r2][t][j] + bv;
      }
    }
}

// ------------------- fused KIVI fake-quant + flash attention ----------------
// grid: (SEQ/64, BATCH*HEADS), block: 128 (4 waves, 16 query rows each).
__global__ __launch_bounds__(128) void kivi_flash_attn(
    const float* __restrict__ qkv,    // [4096][3072] fp32 (Q|K|V)
    const float* __restrict__ mask,   // [BATCH][SEQ] additive
    _Float16* __restrict__ ctx)       // [4096][1024] f16 context out
{
  __shared__ __align__(16) _Float16 lK[64][72];        // dequant keys [key][d]
  __shared__ __align__(16) _Float16 lV[64][72];        // values transposed [d][key]
  __shared__ __align__(16) _Float16 lP[4][16][72];     // per-wave P staging

  const int tid  = threadIdx.x;
  const int wave = tid >> 5;
  const int lane = tid & 31;
  const int ln   = lane & 15;
  const int hf   = (lane < 16) ? 0 : 1;
  const int bh   = blockIdx.y;
  const int b    = bh >> 4;          // HEADS == 16
  const int h    = bh & 15;
  const int q0   = blockIdx.x * 64;

  // Q fragments (A layout), held in registers for the whole kernel.
  const float* qrow = qkv + (size_t)(b * SEQ + q0 + wave * 16 + ln) * (3 * EMB) + h * HD;
  v16h aq[2];
#pragma unroll
  for (int ks = 0; ks < 2; ++ks) {
    const int d0 = ks * 32 + hf * 8;
#pragma unroll
    for (int i = 0; i < 8; ++i) {
      aq[ks][i]     = (_Float16)qrow[d0 + i];
      aq[ks][i + 8] = (_Float16)qrow[d0 + 16 + i];
    }
  }

  v8f o[4];
  float m_i[8], l_i[8];
#pragma unroll
  for (int t = 0; t < 4; ++t)
#pragma unroll
    for (int j = 0; j < 8; ++j) o[t][j] = 0.0f;
#pragma unroll
  for (int j = 0; j < 8; ++j) { m_i[j] = -1e30f; l_i[j] = 0.0f; }

  for (int kb = 0; kb < SEQ / 64; ++kb) {
    __syncthreads();   // previous iteration done with lK/lV/lP

    // Stage 64 keys (KIVI 2-bit fake-quant, groups of 4 along d) and 64 values.
#pragma unroll
    for (int it = 0; it < 8; ++it) {
      const int g   = tid + it * 128;          // 0..1023 groups
      const int key = g >> 4;
      const int grp = g & 15;
      const size_t rowoff =
          (size_t)(b * SEQ + kb * 64 + key) * (3 * EMB) + h * HD + grp * 4;

      const float4 kv = *reinterpret_cast<const float4*>(&qkv[rowoff + EMB]);
      const float ma = fmaxf(fmaxf(fabsf(kv.x), fabsf(kv.y)),
                             fmaxf(fabsf(kv.z), fabsf(kv.w)));
      const float scale = ma * (1.0f / 1.5f);
      const float inv   = (scale == 0.0f) ? 1.0f : (1.0f / scale);
      v4h kq;
      {
        const float c0 = fminf(fmaxf(roundf(kv.x * inv + 1.5f), 0.0f), 3.0f);
        const float c1 = fminf(fmaxf(roundf(kv.y * inv + 1.5f), 0.0f), 3.0f);
        const float c2 = fminf(fmaxf(roundf(kv.z * inv + 1.5f), 0.0f), 3.0f);
        const float c3 = fminf(fmaxf(roundf(kv.w * inv + 1.5f), 0.0f), 3.0f);
        kq[0] = (_Float16)((c0 - 1.5f) * scale);
        kq[1] = (_Float16)((c1 - 1.5f) * scale);
        kq[2] = (_Float16)((c2 - 1.5f) * scale);
        kq[3] = (_Float16)((c3 - 1.5f) * scale);
      }
      *reinterpret_cast<v4h*>(&lK[key][grp * 4]) = kq;

      const float4 vv = *reinterpret_cast<const float4*>(&qkv[rowoff + 2 * EMB]);
      lV[grp * 4 + 0][key] = (_Float16)vv.x;
      lV[grp * 4 + 1][key] = (_Float16)vv.y;
      lV[grp * 4 + 2][key] = (_Float16)vv.z;
      lV[grp * 4 + 3][key] = (_Float16)vv.w;
    }
    __syncthreads();

    // S = Q @ Kdeq^T : 16 q rows x 64 keys per wave (4 tiles, 2 K-steps of 32).
    v8f s[4];
#pragma unroll
    for (int t = 0; t < 4; ++t)
#pragma unroll
      for (int j = 0; j < 8; ++j) s[t][j] = 0.0f;
#pragma unroll
    for (int ks = 0; ks < 2; ++ks) {
#pragma unroll
      for (int t = 0; t < 4; ++t) {
        const int key   = t * 16 + ln;
        const int dbase = ks * 32 + hf * 8;
        const v8h b_lo = *reinterpret_cast<const v8h*>(&lK[key][dbase]);
        const v8h b_hi = *reinterpret_cast<const v8h*>(&lK[key][dbase + 16]);
        s[t] = __builtin_amdgcn_wmma_f32_16x16x32_f16(
            false, aq[ks], false, wmma_pack(b_lo, b_hi),
            (short)0, s[t], false, false);
      }
    }

    // mask BEFORE 1/sqrt(64) scaling (matches reference).
#pragma unroll
    for (int t = 0; t < 4; ++t) {
      const float mk = mask[b * SEQ + kb * 64 + t * 16 + ln];
#pragma unroll
      for (int j = 0; j < 8; ++j) s[t][j] = (s[t][j] + mk) * 0.125f;
    }

    // Online softmax: rows live as (VGPR j, lane-half); reduce across 16 lanes.
    float corr[8];
#pragma unroll
    for (int j = 0; j < 8; ++j) {
      float bm = fmaxf(fmaxf(s[0][j], s[1][j]), fmaxf(s[2][j], s[3][j]));
#pragma unroll
      for (int off = 1; off < 16; off <<= 1) bm = fmaxf(bm, __shfl_xor(bm, off, 16));
      const float mnew = fmaxf(m_i[j], bm);
      corr[j] = __expf(m_i[j] - mnew);
      m_i[j] = mnew;
      float rs = 0.0f;
#pragma unroll
      for (int t = 0; t < 4; ++t) {
        const float p = __expf(s[t][j] - mnew);
        s[t][j] = p;
        rs += p;
      }
#pragma unroll
      for (int off = 1; off < 16; off <<= 1) rs += __shfl_xor(rs, off, 16);
      l_i[j] = l_i[j] * corr[j] + rs;
    }
#pragma unroll
    for (int t = 0; t < 4; ++t)
#pragma unroll
      for (int j = 0; j < 8; ++j) o[t][j] *= corr[j];

    // Stage P (C layout -> LDS) so it can be re-read in A-fragment layout.
#pragma unroll
    for (int t = 0; t < 4; ++t)
#pragma unroll
      for (int j = 0; j < 8; ++j)
        lP[wave][j + hf * 8][t * 16 + ln] = (_Float16)s[t][j];
    __syncthreads();

    // O += P @ V  (K dim = keys, 2 steps of 32; B = V^T staged as lV[d][key]).
#pragma unroll
    for (int ks = 0; ks < 2; ++ks) {
      const int kk = ks * 32 + hf * 8;
      const v8h p_lo = *reinterpret_cast<const v8h*>(&lP[wave][ln][kk]);
      const v8h p_hi = *reinterpret_cast<const v8h*>(&lP[wave][ln][kk + 16]);
      const v16h ap = wmma_pack(p_lo, p_hi);
#pragma unroll
      for (int t = 0; t < 4; ++t) {
        const int dc    = t * 16 + ln;
        const int kbase = ks * 32 + hf * 8;
        const v8h b_lo = *reinterpret_cast<const v8h*>(&lV[dc][kbase]);
        const v8h b_hi = *reinterpret_cast<const v8h*>(&lV[dc][kbase + 16]);
        o[t] = __builtin_amdgcn_wmma_f32_16x16x32_f16(
            false, ap, false, wmma_pack(b_lo, b_hi),
            (short)0, o[t], false, false);
      }
    }
  }

  // Normalize and write f16 context [B,S,H,d] flattened to [4096][1024].
#pragma unroll
  for (int j = 0; j < 8; ++j) {
    const float inv = 1.0f / l_i[j];
    const int row = q0 + wave * 16 + j + hf * 8;
#pragma unroll
    for (int t = 0; t < 4; ++t) {
      ctx[(size_t)(b * SEQ + row) * EMB + h * HD + t * 16 + ln] =
          (_Float16)(o[t][j] * inv);
    }
  }
}

// --------------------------------- launch ----------------------------------
extern "C" void kernel_launch(void* const* d_in, const int* in_sizes, int n_in,
                              void* d_out, int out_size, void* d_ws, size_t ws_size,
                              hipStream_t stream) {
  const float* hs   = (const float*)d_in[0];   // [4,1024,1024]
  const float* mask = (const float*)d_in[1];   // [4,1024]
  const float* wat  = (const float*)d_in[2];   // [1024,3072]
  const float* bat  = (const float*)d_in[3];   // [3072]
  const float* wpr  = (const float*)d_in[4];   // [1024,1024]
  const float* bpr  = (const float*)d_in[5];   // [1024]
  float* out = (float*)d_out;                  // [4,1024,1024]

  // Workspace layout (all 16B-aligned): Xh | Wah | Wph | QKV(f32) | ctx(f16)
  _Float16* Xh  = (_Float16*)d_ws;
  _Float16* Wah = Xh  + (size_t)MTOT * EMB;
  _Float16* Wph = Wah + (size_t)EMB * 3 * EMB;
  float*    QKV = (float*)(Wph + (size_t)EMB * EMB);
  _Float16* ctx = (_Float16*)(QKV + (size_t)MTOT * 3 * EMB);

  const int n_hs = MTOT * EMB / 4;
  const int n_wa = EMB * 3 * EMB / 4;
  const int n_wp = EMB * EMB / 4;
  cvt4<<<(n_hs + 255) / 256, 256, 0, stream>>>(hs,  Xh,  n_hs);
  cvt4<<<(n_wa + 255) / 256, 256, 0, stream>>>(wat, Wah, n_wa);
  cvt4<<<(n_wp + 255) / 256, 256, 0, stream>>>(wpr, Wph, n_wp);

  // QKV = X @ Wattn + b    [4096 x 3072]
  gemm_f16_wmma<<<dim3(3 * EMB / 64, MTOT / 256), 256, 0, stream>>>(
      Xh, Wah, bat, QKV, MTOT, 3 * EMB, EMB);

  // Fused fake-quant + flash attention -> f16 context
  kivi_flash_attn<<<dim3(SEQ / 64, BATCH * HEADS), 128, 0, stream>>>(
      QKV, mask, ctx);

  // out = ctx @ Wproj + b  [4096 x 1024]
  gemm_f16_wmma<<<dim3(EMB / 64, MTOT / 256), 256, 0, stream>>>(
      ctx, Wph, bpr, out, MTOT, EMB, EMB);
}
